// GNN_5927054868943
// MI455X (gfx1250) — matrix operand order repaired and checked by reading, not compiled, not run on hardware
//
#include <hip/hip_runtime.h>
#include <hip/hip_bf16.h>
#include <math.h>

#define GNUM 128
#define EPSN 1e-5f
#define MINF_MAPPED 0x007FFFFFu   // f2u(-inf)

typedef __attribute__((ext_vector_type(16))) __bf16 v16bf;
typedef __attribute__((ext_vector_type(8)))  __bf16 v8bf;
typedef __attribute__((ext_vector_type(8)))  float  v8f;

#define CEILDIV(a,b) (((a)+(b)-1)/(b))

// ---------- helpers ----------
__device__ __forceinline__ unsigned f2u(float f) {
  unsigned u = __float_as_uint(f);
  return (u & 0x80000000u) ? ~u : (u | 0x80000000u);
}
__device__ __forceinline__ float u2f(unsigned u) {
  u = (u & 0x80000000u) ? (u & 0x7fffffffu) : ~u;
  return __uint_as_float(u);
}
__device__ __forceinline__ void atomAddF(float* a, float v) {
  __hip_atomic_fetch_add(a, v, __ATOMIC_RELAXED, __HIP_MEMORY_SCOPE_AGENT);
}
__device__ __forceinline__ void atomMaxU(unsigned* a, unsigned v) {
  __hip_atomic_fetch_max(a, v, __ATOMIC_RELAXED, __HIP_MEMORY_SCOPE_AGENT);
}

// ---------- fills ----------
__global__ void fill_f32(float* p, float v, size_t n) {
  size_t i = (size_t)blockIdx.x * blockDim.x + threadIdx.x;
  if (i < n) p[i] = v;
}
__global__ void fill_u32(unsigned* p, unsigned v, size_t n) {
  size_t i = (size_t)blockIdx.x * blockDim.x + threadIdx.x;
  if (i < n) p[i] = v;
}

// ---------- bf16 WMMA GEMM: C[M,N] = A[M,K] @ W[K,N], fp32 in/out ----------
// Requires M%64==0, N%64==0, K%32==0 (true for all layers with K>1).
// block = 256 thr (8 waves). Block tile 64(M) x 64(N); wave (w>>2) selects
// row group {0,16}, each wave also does row group +32 (2 accumulators
// sharing one B fragment); (w&3) selects 16-col group. A (64x32) and B
// (32x64, col-major) staged in LDS as *bf16* (converted once at staging).
#define AST 40   // A LDS row stride in bf16 elems (80B -> conflict-free)
#define BST 40   // B LDS col stride in bf16 elems
__global__ __launch_bounds__(256)
void gemm_bf16_wmma(const float* __restrict__ A, const float* __restrict__ W,
                    float* __restrict__ C, int M, int N, int K) {
  __shared__ __bf16 As[64 * AST];
  __shared__ __bf16 Bs[64 * BST];
  const int tid  = threadIdx.x;
  const int lane = tid & 31, wave = tid >> 5;
  const int half = lane >> 4, l15 = lane & 15;
  const int wrow = (wave >> 2) << 4;      // 0 | 16  (second acc adds +32)
  const int wcol = (wave & 3) << 4;       // 0,16,32,48
  const int rowBase = blockIdx.x * 64;
  const int colBase = blockIdx.y * 64;

  v8f acc0 = {0.f,0.f,0.f,0.f,0.f,0.f,0.f,0.f};
  v8f acc1 = {0.f,0.f,0.f,0.f,0.f,0.f,0.f,0.f};

  // A staging coords: thread covers row ar, k ak..ak+7
  const int ar = tid >> 2;                // 0..63
  const int ak = (tid & 3) << 3;          // 0,8,16,24
  const float* Ag = A + (size_t)(rowBase + ar) * K + ak;
  // B staging coords: thread covers col bc, k bk..bk+7 (column gather)
  const int bc = tid & 63;                // 0..63
  const int bk = (tid >> 6) << 3;         // 0,8,16,24

  for (int k0 = 0; k0 < K; k0 += 32) {
    // ---- stage A (64x32), convert to bf16, one 16B LDS store/thread ----
    {
      const float4 a0 = *(const float4*)(Ag + k0);
      const float4 a1 = *(const float4*)(Ag + k0 + 4);
      union { v8bf v; uint4 q; } pk;
      pk.v[0] = (__bf16)a0.x; pk.v[1] = (__bf16)a0.y;
      pk.v[2] = (__bf16)a0.z; pk.v[3] = (__bf16)a0.w;
      pk.v[4] = (__bf16)a1.x; pk.v[5] = (__bf16)a1.y;
      pk.v[6] = (__bf16)a1.z; pk.v[7] = (__bf16)a1.w;
      *(uint4*)&As[ar * AST + ak] = pk.q;
    }
    // ---- stage B (32x64) col-major: 8 coalesced b32 loads -> 16B store ----
    {
      const float* wp = W + (size_t)(k0 + bk) * N + colBase + bc;
      union { v8bf v; uint4 q; } pk;
#pragma unroll
      for (int i = 0; i < 8; ++i) pk.v[i] = (__bf16)wp[(size_t)i * N];
      *(uint4*)&Bs[bc * BST + bk] = pk.q;
    }
    if (k0 + 32 < K) {
      __builtin_prefetch(Ag + k0 + 32, 0, 3);
      __builtin_prefetch(W + (size_t)(k0 + 32 + bk) * N + colBase + bc, 0, 3);
    }
    __syncthreads();

    // ---- fragments: pure 16B ds loads (CDNA5 wave32 layouts) ----
    union { v16bf v; uint4 q[2]; } af0, af1, bfm;
    const __bf16* a0p = &As[(wrow + l15) * AST + half * 8];
    af0.q[0] = *(const uint4*)(a0p);          // K  h*8 .. h*8+7
    af0.q[1] = *(const uint4*)(a0p + 16);     // K  h*8+16 .. h*8+23
    const __bf16* a1p = a0p + 32 * AST;       // rows +32
    af1.q[0] = *(const uint4*)(a1p);
    af1.q[1] = *(const uint4*)(a1p + 16);
    const __bf16* bp = &Bs[(wcol + l15) * BST + half * 8];
    bfm.q[0] = *(const uint4*)(bp);           // K h*8 .. h*8+7
    bfm.q[1] = *(const uint4*)(bp + 16);      // K h*8+16 .. h*8+23

    acc0 = __builtin_amdgcn_wmma_f32_16x16x32_bf16(false, af0.v, false, bfm.v,
                                                   (short)0, acc0, false, false);
    acc1 = __builtin_amdgcn_wmma_f32_16x16x32_bf16(false, af1.v, false, bfm.v,
                                                   (short)0, acc1, false, false);
    __syncthreads();
  }

  const int col = colBase + wcol + l15;
#pragma unroll
  for (int r = 0; r < 8; ++r) {
    C[(size_t)(rowBase + wrow + r + half * 8) * N + col] = acc0[r];
    C[(size_t)(rowBase + wrow + r + half * 8 + 32) * N + col] = acc1[r];
  }
}

// ---------- K=1 "GEMM" (layer 1): outer product H[n,f] = x[n] * W[0,f] ----------
__global__ void gemm_k1(const float* __restrict__ x, const float* __restrict__ W,
                        float* __restrict__ H, size_t nf, int fsh) {
  size_t i = (size_t)blockIdx.x * blockDim.x + threadIdx.x;
  if (i >= nf) return;
  int n = (int)(i >> fsh), f = (int)(i & ((((size_t)1) << fsh) - 1));
  H[i] = x[n] * W[f];
}

// ---------- per-node attention dots: es=h@a_s, ed=h@a_d (one wave per node) ----------
__global__ void node_dots(const float* __restrict__ H, const float* __restrict__ as_,
                          const float* __restrict__ ad_, float* __restrict__ es,
                          float* __restrict__ ed, int n, int F) {
  int gw = (int)((blockIdx.x * (size_t)blockDim.x + threadIdx.x) >> 5);
  int lane = threadIdx.x & 31;
  if (gw >= n) return;
  const float* h = H + (size_t)gw * F;
  float s = 0.f, d = 0.f;
  for (int f = lane; f < F; f += 32) {
    float hv = h[f];
    s += hv * as_[f];
    d += hv * ad_[f];
  }
#pragma unroll
  for (int o = 16; o; o >>= 1) {
    s += __shfl_xor(s, o, 32);
    d += __shfl_xor(d, o, 32);
  }
  if (lane == 0) { es[gw] = s; ed[gw] = d; }
}

// ---------- GAT edge softmax ----------
__global__ void edge_logits(const int* __restrict__ src, const int* __restrict__ dst,
                            const float* __restrict__ es, const float* __restrict__ ed,
                            float* __restrict__ p, unsigned* __restrict__ m,
                            int E_, int E2_) {
  int e = blockIdx.x * blockDim.x + threadIdx.x;
  if (e >= E2_) return;
  int s = (e < E_) ? src[e] : (e - E_);
  int d = (e < E_) ? dst[e] : (e - E_);
  float v = es[s] + ed[d];
  v = (v > 0.f) ? v : 0.2f * v;          // leaky_relu(0.2)
  p[e] = v;
  atomMaxU(m + d, f2u(v));
}

__global__ void edge_exp(const int* __restrict__ dst, float* __restrict__ p,
                         const unsigned* __restrict__ m, float* __restrict__ z,
                         int E_, int E2_) {
  int e = blockIdx.x * blockDim.x + threadIdx.x;
  if (e >= E2_) return;
  int d = (e < E_) ? dst[e] : (e - E_);
  float v = __expf(p[e] - u2f(m[d]));
  p[e] = v;
  atomAddF(z + d, v);
}

__global__ void edge_div(const int* __restrict__ dst, float* __restrict__ p,
                         const float* __restrict__ z, int E_, int E2_) {
  int e = blockIdx.x * blockDim.x + threadIdx.x;
  if (e >= E2_) return;
  int d = (e < E_) ? dst[e] : (e - E_);
  p[e] = p[e] / z[d];
}

// ---------- feature scatters (F is a power of two; fsh = log2(F)) ----------
__global__ void scatter_gat(const int* __restrict__ src, const int* __restrict__ dst,
                            const float* __restrict__ coef, const float* __restrict__ H,
                            float* __restrict__ O, int E_, int E2_, int fsh) {
  size_t i = (size_t)blockIdx.x * blockDim.x + threadIdx.x;
  size_t total = ((size_t)E2_) << fsh;
  if (i >= total) return;
  int e = (int)(i >> fsh);
  int f = (int)(i & ((((size_t)1) << fsh) - 1));
  int s = (e < E_) ? src[e] : (e - E_);
  int d = (e < E_) ? dst[e] : (e - E_);
  atomAddF(&O[(((size_t)d) << fsh) + f], coef[e] * H[(((size_t)s) << fsh) + f]);
}

__global__ void scatter_gcn(const int* __restrict__ src, const int* __restrict__ dst,
                            const float* __restrict__ dis, const float* __restrict__ H,
                            float* __restrict__ O, int E_, int E2_, int fsh) {
  size_t i = (size_t)blockIdx.x * blockDim.x + threadIdx.x;
  size_t total = ((size_t)E2_) << fsh;
  if (i >= total) return;
  int e = (int)(i >> fsh);
  int f = (int)(i & ((((size_t)1) << fsh) - 1));
  int s = (e < E_) ? src[e] : (e - E_);
  int d = (e < E_) ? dst[e] : (e - E_);
  atomAddF(&O[(((size_t)d) << fsh) + f], dis[s] * dis[d] * H[(((size_t)s) << fsh) + f]);
}

// ---------- bias + ELU ----------
__global__ void bias_elu(float* __restrict__ O, const float* __restrict__ b,
                         size_t nf, int fmask) {
  size_t i = (size_t)blockIdx.x * blockDim.x + threadIdx.x;
  if (i >= nf) return;
  float v = O[i] + b[i & (size_t)fmask];
  O[i] = (v > 0.f) ? v : (__expf(v) - 1.f);
}

// ---------- GraphNorm (3 passes) ----------
__global__ void gn_sum(const float* __restrict__ X, const int* __restrict__ batch,
                       float* __restrict__ gsum, size_t nf, int fsh, int F) {
  size_t i = (size_t)blockIdx.x * blockDim.x + threadIdx.x;
  if (i >= nf) return;
  int n = (int)(i >> fsh), f = (int)(i & (size_t)(F - 1));
  atomAddF(&gsum[(size_t)batch[n] * F + f], X[i]);
}
__global__ void gn_var(const float* __restrict__ X, const int* __restrict__ batch,
                       const float* __restrict__ gsum, const float* __restrict__ cnt,
                       const float* __restrict__ alpha, float* __restrict__ gsq,
                       size_t nf, int fsh, int F) {
  size_t i = (size_t)blockIdx.x * blockDim.x + threadIdx.x;
  if (i >= nf) return;
  int n = (int)(i >> fsh), f = (int)(i & (size_t)(F - 1));
  int g = batch[n];
  float c = fmaxf(cnt[g], 1.f);
  float sub = X[i] - alpha[f] * (gsum[(size_t)g * F + f] / c);
  atomAddF(&gsq[(size_t)g * F + f], sub * sub);
}
__global__ void gn_fin(const float* __restrict__ X, const int* __restrict__ batch,
                       const float* __restrict__ gsum, const float* __restrict__ gsq,
                       const float* __restrict__ cnt, const float* __restrict__ alpha,
                       const float* __restrict__ gamma, const float* __restrict__ beta,
                       float* __restrict__ Y, size_t nf, int fsh, int F) {
  size_t i = (size_t)blockIdx.x * blockDim.x + threadIdx.x;
  if (i >= nf) return;
  int n = (int)(i >> fsh), f = (int)(i & (size_t)(F - 1));
  int g = batch[n];
  float c = fmaxf(cnt[g], 1.f);
  float sub = X[i] - alpha[f] * (gsum[(size_t)g * F + f] / c);
  float var = gsq[(size_t)g * F + f] / c;
  Y[i] = gamma[f] * sub * rsqrtf(var + EPSN) + beta[f];
}

// ---------- degree / counts ----------
__global__ void deg_edges(const int* __restrict__ dst, float* __restrict__ deg, int E_) {
  int e = blockIdx.x * blockDim.x + threadIdx.x;
  if (e < E_) atomAddF(&deg[dst[e]], 1.f);
}
__global__ void rsqrt_k(const float* __restrict__ deg, float* __restrict__ dis, int n) {
  int i = blockIdx.x * blockDim.x + threadIdx.x;
  if (i < n) dis[i] = rsqrtf(deg[i]);
}
__global__ void count_nodes(const int* __restrict__ batch, float* __restrict__ cntG, int n) {
  int i = blockIdx.x * blockDim.x + threadIdx.x;
  if (i < n) atomAddF(&cntG[batch[i]], 1.f);
}

// ---------- pooling + head ----------
__global__ void pool_k(const float* __restrict__ X, const int* __restrict__ batch,
                       float* __restrict__ psum, unsigned* __restrict__ pmax, size_t nf) {
  size_t i = (size_t)blockIdx.x * blockDim.x + threadIdx.x;
  if (i >= nf) return;
  int n = (int)(i >> 6), f = (int)(i & 63);
  int g = batch[n];
  float v = X[i];
  atomAddF(&psum[(size_t)g * 64 + f], v);
  atomMaxU(&pmax[(size_t)g * 64 + f], f2u(v));
}
__global__ void final_linear(const unsigned* __restrict__ pmax,
                             const float* __restrict__ psum,
                             const float* __restrict__ cntG,
                             const float* __restrict__ lW, const float* __restrict__ lb,
                             float* __restrict__ out) {
  int tid = threadIdx.x;
  if (tid >= GNUM * 2) return;
  int g = tid >> 1, j = tid & 1;
  float c = fmaxf(cntG[g], 1.f);
  float acc = lb[j];
  for (int f = 0; f < 64; ++f) {
    float mx = u2f(pmax[g * 64 + f]);
    if (!isfinite(mx)) mx = 0.f;         // empty graphs -> 0
    float sm = psum[g * 64 + f];
    acc += mx * lW[f * 2 + j] + (sm / c) * lW[(64 + f) * 2 + j] + sm * lW[(128 + f) * 2 + j];
  }
  out[g * 2 + j] = acc;
}

// ---------- launcher ----------
extern "C" void kernel_launch(void* const* d_in, const int* in_sizes, int n_in,
                              void* d_out, int out_size, void* d_ws, size_t ws_size,
                              hipStream_t stream) {
  const int N  = in_sizes[0];
  const int E  = in_sizes[1] / 2;
  const int E2 = E + N;

  const float* x    = (const float*)d_in[0];
  const int*  src   = (const int*)d_in[1];
  const int*  dst   = src + E;
  const int*  batch = (const int*)d_in[2];

  const float* gW1 = (const float*)d_in[3],  *gas1 = (const float*)d_in[4],
             *gad1 = (const float*)d_in[5],  *gb1  = (const float*)d_in[6];
  const float* gW2 = (const float*)d_in[7],  *gas2 = (const float*)d_in[8],
             *gad2 = (const float*)d_in[9],  *gb2  = (const float*)d_in[10];
  const float* gW3 = (const float*)d_in[11], *gas3 = (const float*)d_in[12],
             *gad3 = (const float*)d_in[13], *gb3  = (const float*)d_in[14];
  const float* cW1 = (const float*)d_in[15], *cb1 = (const float*)d_in[16];
  const float* cW2 = (const float*)d_in[17], *cb2 = (const float*)d_in[18];
  const float* nw[5], *nb[5], *na[5];
  for (int i = 0; i < 5; ++i) {
    nw[i] = (const float*)d_in[19 + 3 * i];
    nb[i] = (const float*)d_in[20 + 3 * i];
    na[i] = (const float*)d_in[21 + 3 * i];
  }
  const float* lW = (const float*)d_in[34];
  const float* lb = (const float*)d_in[35];

  // workspace bump allocator
  char* wp = (char*)d_ws;
  auto alloc = [&](size_t bytes) -> void* {
    void* r = (void*)wp;
    wp += (bytes + 255) & ~(size_t)255;
    return r;
  };
  const size_t NF = (size_t)N * 2048;
  float*    bufX = (float*)alloc(NF * 4);
  float*    bufH = (float*)alloc(NF * 4);
  float*    bufO = (float*)alloc(NF * 4);
  float*    es   = (float*)alloc((size_t)N * 4);
  float*    ed   = (float*)alloc((size_t)N * 4);
  unsigned* mU   = (unsigned*)alloc((size_t)N * 4);
  float*    z    = (float*)alloc((size_t)N * 4);
  float*    pC   = (float*)alloc((size_t)E2 * 4);
  float*    deg  = (float*)alloc((size_t)N * 4);
  float*    dis  = (float*)alloc((size_t)N * 4);
  float*    cntG = (float*)alloc((size_t)GNUM * 4);
  float*    gsum = (float*)alloc((size_t)GNUM * 2048 * 4);
  float*    gsq  = (float*)alloc((size_t)GNUM * 2048 * 4);
  unsigned* pmax = (unsigned*)alloc((size_t)GNUM * 64 * 4);
  float*    psum = (float*)alloc((size_t)GNUM * 64 * 4);

  auto fillF = [&](float* ptr, float v, size_t n) {
    fill_f32<<<(unsigned)CEILDIV(n, (size_t)256), 256, 0, stream>>>(ptr, v, n);
  };
  auto fillU = [&](unsigned* ptr, unsigned v, size_t n) {
    fill_u32<<<(unsigned)CEILDIV(n, (size_t)256), 256, 0, stream>>>(ptr, v, n);
  };

  auto run_gnorm = [&](const float* Xin, float* Xout, const float* gm,
                       const float* bt, const float* al, int F, int fsh) {
    size_t gf = (size_t)GNUM * F, nf = (size_t)N * F;
    fillF(gsum, 0.f, gf);
    fillF(gsq, 0.f, gf);
    unsigned blks = (unsigned)CEILDIV(nf, (size_t)256);
    gn_sum<<<blks, 256, 0, stream>>>(Xin, batch, gsum, nf, fsh, F);
    gn_var<<<blks, 256, 0, stream>>>(Xin, batch, gsum, cntG, al, gsq, nf, fsh, F);
    gn_fin<<<blks, 256, 0, stream>>>(Xin, batch, gsum, gsq, cntG, al, gm, bt, Xout, nf, fsh, F);
  };

  auto run_gat = [&](const float* Xin, int K, int F, const float* W,
                     const float* as_, const float* ad_, const float* b,
                     const float* gm, const float* bt, const float* al) {
    int fsh = __builtin_ctz(F);
    size_t nf = (size_t)N * F, ef = (size_t)E2 << fsh;
    if (K == 1) {
      gemm_k1<<<(unsigned)CEILDIV(nf, (size_t)256), 256, 0, stream>>>(Xin, W, bufH, nf, fsh);
    } else {
      dim3 gg(N / 64, F / 64);
      gemm_bf16_wmma<<<gg, 256, 0, stream>>>(Xin, W, bufH, N, F, K);
    }
    node_dots<<<CEILDIV(N, 8), 256, 0, stream>>>(bufH, as_, ad_, es, ed, N, F);
    fillU(mU, MINF_MAPPED, (size_t)N);
    fillF(z, 0.f, (size_t)N);
    unsigned eb = (unsigned)CEILDIV(E2, 256);
    edge_logits<<<eb, 256, 0, stream>>>(src, dst, es, ed, pC, mU, E, E2);
    edge_exp<<<eb, 256, 0, stream>>>(dst, pC, mU, z, E, E2);
    edge_div<<<eb, 256, 0, stream>>>(dst, pC, z, E, E2);
    fillF(bufO, 0.f, nf);
    scatter_gat<<<(unsigned)CEILDIV(ef, (size_t)256), 256, 0, stream>>>(
        src, dst, pC, bufH, bufO, E, E2, fsh);
    bias_elu<<<(unsigned)CEILDIV(nf, (size_t)256), 256, 0, stream>>>(bufO, b, nf, F - 1);
    run_gnorm(bufO, bufX, gm, bt, al, F, fsh);
  };

  auto run_gcn = [&](const float* Xin, int K, int F, const float* W, const float* b,
                     const float* gm, const float* bt, const float* al) {
    int fsh = __builtin_ctz(F);
    size_t nf = (size_t)N * F, ef = (size_t)E2 << fsh;
    dim3 gg(N / 64, F / 64);
    gemm_bf16_wmma<<<gg, 256, 0, stream>>>(Xin, W, bufH, N, F, K);
    fillF(bufO, 0.f, nf);
    scatter_gcn<<<(unsigned)CEILDIV(ef, (size_t)256), 256, 0, stream>>>(
        src, dst, dis, bufH, bufO, E, E2, fsh);
    bias_elu<<<(unsigned)CEILDIV(nf, (size_t)256), 256, 0, stream>>>(bufO, b, nf, F - 1);
    run_gnorm(bufO, bufX, gm, bt, al, F, fsh);
  };

  // graph-constant stats (recomputed each call; ws not persistent)
  fillF(deg, 1.f, (size_t)N);                                   // self loops
  deg_edges<<<(unsigned)CEILDIV(E, 256), 256, 0, stream>>>(dst, deg, E);
  rsqrt_k<<<(unsigned)CEILDIV(N, 256), 256, 0, stream>>>(deg, dis, N);
  fillF(cntG, 0.f, (size_t)GNUM);
  count_nodes<<<(unsigned)CEILDIV(N, 256), 256, 0, stream>>>(batch, cntG, N);

  // 5-layer pipeline
  run_gat(x,    1,   64,   gW1, gas1, gad1, gb1, nw[0], nb[0], na[0]);
  run_gat(bufX, 64,  512,  gW2, gas2, gad2, gb2, nw[1], nb[1], na[1]);
  run_gat(bufX, 512, 2048, gW3, gas3, gad3, gb3, nw[2], nb[2], na[2]);
  run_gcn(bufX, 2048, 512, cW1, cb1, nw[3], nb[3], na[3]);
  run_gcn(bufX, 512,  64,  cW2, cb2, nw[4], nb[4], na[4]);

  // pooling + head
  fillU(pmax, MINF_MAPPED, (size_t)GNUM * 64);
  fillF(psum, 0.f, (size_t)GNUM * 64);
  size_t pn = (size_t)N * 64;
  pool_k<<<(unsigned)CEILDIV(pn, (size_t)256), 256, 0, stream>>>(bufX, batch, psum, pmax, pn);
  final_linear<<<1, 256, 0, stream>>>(pmax, psum, cntG, lW, lb, (float*)d_out);
}
